// NerfGLUBlock_63823214019112
// MI455X (gfx1250) — matrix operand order repaired and checked by reading, not compiled, not run on hardware
//
#include <hip/hip_runtime.h>
#include <hip/hip_bf16.h>

// ---------------- problem constants ----------------
#define D_    64
#define H_    256            // D * MLP_RATIO
#define S_    1024
#define NTOK  4096
#define B_    64
#define MAT   16384          // D_ * H_
#define TOTAL 49152          // 3 * MAT

typedef __attribute__((ext_vector_type(2)))  float  v2f;
typedef __attribute__((ext_vector_type(8)))  float  v8f;
typedef __attribute__((ext_vector_type(8)))  __bf16 v8bf;
typedef __attribute__((ext_vector_type(16))) __bf16 v16bf;

union V16 { v16bf v; v8bf h[2]; };

// Swizzled address of element (k,n) of a KxN weight matrix stored as a stream
// of 16x16x32 B-fragments (contiguous k order, shared with the A loads):
//   frag  = (k/32)*ntiles + n/16           (512 bf16 each)
//   lane  = ((k%32)/16)*16 + n%16          (lanes 0-15: K 0-15, 16-31: K 16-31)
//   elem  = k%16                           (VGPR j holds K=2j,2j+1 packed)
__device__ __forceinline__ int swz(int k, int n, int ntiles) {
  return (((k >> 5) * ntiles + (n >> 4)) << 9)
       + (((((k >> 4) & 1) << 4) + (n & 15)) << 4)
       + (k & 15);
}

// =====================================================================
// Kernel A: mlp_params = s @ W_pg + b_pg      (f32 WMMA 16x16x4)
// Each wave: 64(M) x 16(N) stripe, K=1024. W_pg (201 MB) read exactly once.
// =====================================================================
__global__ __launch_bounds__(256) void hyper_gemm_kernel(
    const float* __restrict__ s, const float* __restrict__ W,
    const float* __restrict__ bias, float* __restrict__ params) {
  const int lane = threadIdx.x & 31;
  const int wave = threadIdx.x >> 5;
  const int lm   = lane & 15;
  const int lh   = lane >> 4;
  const int cb   = (blockIdx.x * 8 + wave) * 16;   // column base

  const v8f vz = {0.f,0.f,0.f,0.f,0.f,0.f,0.f,0.f};
  v8f acc0 = vz, acc1 = vz, acc2 = vz, acc3 = vz;

  #pragma unroll 2
  for (int k0 = 0; k0 < S_; k0 += 4) {
    const int kk = k0 + lh * 2;          // per-lane K pair (A and B share this)
    v2f bm;
    bm.x = W[(size_t)kk       * TOTAL + cb + lm];
    bm.y = W[(size_t)(kk + 1) * TOTAL + cb + lm];
    v2f a0 = *(const v2f*)(s + ( 0 + lm) * S_ + kk);
    v2f a1 = *(const v2f*)(s + (16 + lm) * S_ + kk);
    v2f a2 = *(const v2f*)(s + (32 + lm) * S_ + kk);
    v2f a3 = *(const v2f*)(s + (48 + lm) * S_ + kk);
    acc0 = __builtin_amdgcn_wmma_f32_16x16x4_f32(false, a0, false, bm, (short)0, acc0, false, false);
    acc1 = __builtin_amdgcn_wmma_f32_16x16x4_f32(false, a1, false, bm, (short)0, acc1, false, false);
    acc2 = __builtin_amdgcn_wmma_f32_16x16x4_f32(false, a2, false, bm, (short)0, acc2, false, false);
    acc3 = __builtin_amdgcn_wmma_f32_16x16x4_f32(false, a3, false, bm, (short)0, acc3, false, false);
  }

  const float bv = bias[cb + lm];
  #pragma unroll
  for (int j = 0; j < 8; ++j) {
    const int m = j + lh * 8;            // C/D layout: VGPR j -> rows j, j+8
    params[(size_t)(m +  0) * TOTAL + cb + lm] = acc0[j] + bv;
    params[(size_t)(m + 16) * TOTAL + cb + lm] = acc1[j] + bv;
    params[(size_t)(m + 32) * TOTAL + cb + lm] = acc2[j] + bv;
    params[(size_t)(m + 48) * TOTAL + cb + lm] = acc3[j] + bv;
  }
}

// =====================================================================
// Kernel B: L2-normalize along dim -2, cast to bf16, pre-swizzle into
// WMMA B-fragment order. 36864 threads total (144 blocks x 256).
// =====================================================================
__global__ __launch_bounds__(256) void norm_pack_kernel(
    const float* __restrict__ p, __bf16* __restrict__ wsw) {
  const int t = blockIdx.x * 256 + threadIdx.x;
  if (t < 2 * B_ * H_) {
    // fc1_gate / fc1_value: normalize over d (64, stride H_) for each (b,h)
    const int h  = t & (H_ - 1);
    const int b  = (t >> 8) & (B_ - 1);
    const int wg = t >> 14;                       // 0 = gate, 1 = value
    const float* src = p + (size_t)b * TOTAL + wg * MAT + h;
    float ss = 0.f;
    #pragma unroll 4
    for (int d = 0; d < D_; ++d) { float v = src[(size_t)d * H_]; ss += v * v; }
    const float rn = 1.f / fmaxf(sqrtf(ss), 1e-12f);
    __bf16* dst = wsw + (size_t)b * TOTAL + wg * MAT;
    #pragma unroll 4
    for (int d = 0; d < D_; ++d)
      dst[swz(d, h, H_ / 16)] = (__bf16)(src[(size_t)d * H_] * rn);
  } else {
    // fc2: normalize over h (256, stride D_) for each (b,d)
    const int t2 = t - 2 * B_ * H_;
    const int d = t2 & (D_ - 1);
    const int b = t2 >> 6;
    const float* src = p + (size_t)b * TOTAL + 2 * MAT + d;
    float ss = 0.f;
    #pragma unroll 4
    for (int hh = 0; hh < H_; ++hh) { float v = src[(size_t)hh * D_]; ss += v * v; }
    const float rn = 1.f / fmaxf(sqrtf(ss), 1e-12f);
    __bf16* dst = wsw + (size_t)b * TOTAL + 2 * MAT;
    #pragma unroll 4
    for (int hh = 0; hh < H_; ++hh)
      dst[swz(hh, d, D_ / 16)] = (__bf16)(src[(size_t)hh * D_] * rn);
  }
}

// =====================================================================
// Kernel C: fused RMSNorm -> xn@fc1_{gate,value} -> SiLU*value -> h@fc2
//           -> +residual.  bf16 WMMA 16x16x32, f32 accumulate.
// Block = 256 threads (8 waves), each wave owns a 16-row tile.
// =====================================================================
__global__ __launch_bounds__(256) void fused_glu_kernel(
    const float* __restrict__ x, const float* __restrict__ scale,
    const __bf16* __restrict__ wsw, float* __restrict__ out) {
  __shared__ __bf16 smem[8][16 * H_];   // 64 KB; per-wave scratch: xn then h
  const int lane = threadIdx.x & 31;
  const int wave = threadIdx.x >> 5;
  const int lm   = lane & 15;
  const int lh   = lane >> 4;
  const int b    = blockIdx.y;
  const int r0   = blockIdx.x * 128 + wave * 16;       // row tile base
  const __bf16* wb = wsw + (size_t)b * TOTAL;
  const v8f vz = {0.f,0.f,0.f,0.f,0.f,0.f,0.f,0.f};

  // ---- RMSNorm: row lm, column half lh (32 floats per lane) ----
  const float* xrow = x + ((size_t)b * NTOK + r0 + lm) * D_ + lh * 32;
  float4 xv[8];
  float ss = 0.f;
  #pragma unroll
  for (int j = 0; j < 8; ++j) {
    xv[j] = ((const float4*)xrow)[j];
    ss += xv[j].x*xv[j].x + xv[j].y*xv[j].y + xv[j].z*xv[j].z + xv[j].w*xv[j].w;
  }
  ss += __shfl_xor(ss, 16, 32);                         // combine column halves
  const float rr = rsqrtf(ss * (1.f / 64.f) + 1e-6f);
  __bf16* xn = &smem[wave][0];                          // [16][64] bf16
  #pragma unroll
  for (int j = 0; j < 8; ++j) {
    float4 sc = ((const float4*)(scale + lh * 32))[j];
    const int o = lm * 64 + lh * 32 + j * 4;
    xn[o + 0] = (__bf16)(xv[j].x * rr * sc.x);
    xn[o + 1] = (__bf16)(xv[j].y * rr * sc.y);
    xn[o + 2] = (__bf16)(xv[j].z * rr * sc.z);
    xn[o + 3] = (__bf16)(xv[j].w * rr * sc.w);
  }

  // ---- A fragments for GEMM1: xn 16x64 = two 16x32 slabs ----
  // A layout used here (contiguous): K(e,l) = (l/16)*16 + e  -> two 16B octets
  V16 a1[2];
  #pragma unroll
  for (int kt = 0; kt < 2; ++kt)
    #pragma unroll
    for (int o = 0; o < 2; ++o)
      a1[kt].h[o] = *(const v8bf*)&smem[wave][lm * 64 + kt * 32 + lh * 16 + o * 8];

  // ---- GEMM1 + SwiGLU, h staged into same per-wave LDS region ----
  __bf16* hbuf = &smem[wave][0];                        // [16][256] bf16
  const __bf16* wg = wb;                                // gate  frags
  const __bf16* wv = wb + MAT;                          // value frags
  #pragma unroll 1
  for (int h0 = 0; h0 < 16; ++h0) {
    v8f ag = vz, av = vz;
    #pragma unroll
    for (int kt = 0; kt < 2; ++kt) {
      v16bf bg = *(const v16bf*)(wg + ((kt * 16 + h0) << 9) + lane * 16);
      ag = __builtin_amdgcn_wmma_f32_16x16x32_bf16(false, a1[kt].v, false, bg, (short)0, ag, false, false);
      v16bf bv = *(const v16bf*)(wv + ((kt * 16 + h0) << 9) + lane * 16);
      av = __builtin_amdgcn_wmma_f32_16x16x32_bf16(false, a1[kt].v, false, bv, (short)0, av, false, false);
    }
    #pragma unroll
    for (int j = 0; j < 8; ++j) {
      const float g = ag[j];
      const float hval = (g / (1.f + __expf(-g))) * av[j];
      hbuf[(j + lh * 8) * H_ + h0 * 16 + lm] = (__bf16)hval;   // C-layout scatter
    }
  }

  // ---- GEMM2: out16x64 = h(16x256) @ fc2(256x64), + residual ----
  const __bf16* w2 = wb + 2 * MAT;
  #pragma unroll 1
  for (int nt = 0; nt < 4; ++nt) {
    v8f acc = vz;
    #pragma unroll
    for (int kt = 0; kt < 8; ++kt) {
      V16 a2;
      a2.h[0] = *(const v8bf*)&smem[wave][lm * H_ + kt * 32 +     lh * 16];
      a2.h[1] = *(const v8bf*)&smem[wave][lm * H_ + kt * 32 + 8 + lh * 16];
      v16bf b2 = *(const v16bf*)(w2 + ((kt * 4 + nt) << 9) + lane * 16);
      acc = __builtin_amdgcn_wmma_f32_16x16x32_bf16(false, a2.v, false, b2, (short)0, acc, false, false);
    }
    #pragma unroll
    for (int j = 0; j < 8; ++j) {
      const size_t idx = ((size_t)b * NTOK + r0 + j + lh * 8) * D_ + nt * 16 + lm;
      out[idx] = acc[j] + x[idx];
    }
  }
}

// =====================================================================
extern "C" void kernel_launch(void* const* d_in, const int* in_sizes, int n_in,
                              void* d_out, int out_size, void* d_ws, size_t ws_size,
                              hipStream_t stream) {
  const float* x     = (const float*)d_in[0];
  const float* s     = (const float*)d_in[1];
  const float* W_pg  = (const float*)d_in[2];
  const float* b_pg  = (const float*)d_in[3];
  const float* scale = (const float*)d_in[4];
  float* out = (float*)d_out;

  const size_t params_bytes = (size_t)B_ * TOTAL * sizeof(float);   // 12.6 MB
  const size_t wsw_bytes    = (size_t)B_ * TOTAL * sizeof(__bf16);  //  6.3 MB
  if (ws_size < params_bytes + wsw_bytes) return;
  float*  params = (float*)d_ws;
  __bf16* wsw    = (__bf16*)((char*)d_ws + params_bytes);

  // A: hypernet GEMM (384 blocks x 8 waves x 16 cols = 49152 columns)
  hyper_gemm_kernel<<<dim3(TOTAL / 128), 256, 0, stream>>>(s, W_pg, b_pg, params);
  // B: normalize + bf16 + fragment swizzle (36864 threads)
  norm_pack_kernel<<<dim3((2 * B_ * H_ + B_ * D_) / 256), 256, 0, stream>>>(params, wsw);
  // C: fused main path (32 row-tiles x 64 batches)
  fused_glu_kernel<<<dim3(NTOK / 128, B_), 256, 0, stream>>>(x, scale, wsw, out);
}
